// SigmaMoE_80169859547318
// MI455X (gfx1250) — matrix operand order repaired and checked by reading, not compiled, not run on hardware
//
#include <hip/hip_runtime.h>

// ---------------- problem constants ----------------
constexpr int Bb = 2, Ss = 4096, Dd = 1024, Ee = 16, Ff = 256, Kk = 4;
constexpr int Nn = Bb * Ss;          // 8192 tokens

typedef __attribute__((ext_vector_type(16))) __bf16 v16bf;
typedef __attribute__((ext_vector_type(8)))  __bf16 v8bf;
typedef __attribute__((ext_vector_type(4)))  __bf16 v4bf;
typedef __attribute__((ext_vector_type(8)))  float  v8f;

// s_wait_asynccnt: builtin if available, else raw SOPP
#if defined(__has_builtin) && __has_builtin(__builtin_amdgcn_s_wait_asynccnt)
#define WAIT_ASYNC(n) __builtin_amdgcn_s_wait_asynccnt(n)
#else
#define WAIT_ASYNC(n) asm volatile("s_wait_asynccnt %0" :: "i"(n) : "memory")
#endif

// one 16-byte global -> LDS async copy per lane (ASYNCcnt-tracked)
__device__ __forceinline__ void async_ld16(unsigned ldsoff, const __bf16* g) {
    asm volatile("global_load_async_to_lds_b128 %0, %1, off"
                 :: "v"(ldsoff), "v"(g)
                 : "memory");
}

// ---------------- zero output + counters ----------------
__global__ void zero_init(float4* __restrict__ out4, int* __restrict__ counts, long n4) {
    long i = (long)blockIdx.x * blockDim.x + threadIdx.x;
    if (i < n4) out4[i] = make_float4(0.f, 0.f, 0.f, 0.f);
    if (i < Ee) counts[i] = 0;
}

// ---------------- x fp32 -> bf16 ----------------
__global__ void convert_x(const float4* __restrict__ x4, v4bf* __restrict__ xb4, long n4) {
    long i = (long)blockIdx.x * blockDim.x + threadIdx.x;
    if (i >= n4) return;
    float4 v = x4[i];
    v4bf o;
    o[0] = (__bf16)v.x; o[1] = (__bf16)v.y; o[2] = (__bf16)v.z; o[3] = (__bf16)v.w;
    xb4[i] = o;
}

// ---------------- per-expert transpose+convert: src [R,C] f32 -> dst [C,R] bf16 ----------------
__global__ void transpose_bf16(const float* __restrict__ src, __bf16* __restrict__ dst,
                               int R, int C) {
    __shared__ float tile[32][33];
    const int e = blockIdx.z;
    const float* s = src + (size_t)e * R * C;
    __bf16*      d = dst + (size_t)e * R * C;
    const int c0 = blockIdx.x * 32, r0 = blockIdx.y * 32;
    const int tx = threadIdx.x, ty = threadIdx.y;           // block (32,8)
    #pragma unroll
    for (int j = 0; j < 32; j += 8)
        tile[ty + j][tx] = s[(size_t)(r0 + ty + j) * C + (c0 + tx)];
    __syncthreads();
    #pragma unroll
    for (int j = 0; j < 32; j += 8)
        d[(size_t)(c0 + ty + j) * R + (r0 + tx)] = (__bf16)tile[tx][ty + j];
}

// ---------------- router: scores, sigmoid, top-4, expert bucketing ----------------
__global__ void route_kernel(const float* __restrict__ x, const float* __restrict__ Wsel,
                             int* __restrict__ counts, int* __restrict__ lists,
                             float* __restrict__ glist) {
    const int wave = threadIdx.x >> 5;
    const int lane = threadIdx.x & 31;
    const int t = blockIdx.x * 8 + wave;
    const int e = lane & 15, part = lane >> 4;
    const float* xr = x    + (size_t)t * Dd + part * (Dd / 2);
    const float* wr = Wsel + (size_t)e * Dd + part * (Dd / 2);
    float s = 0.f;
    for (int d = 0; d < Dd / 2; d += 4) {
        float4 xv = *(const float4*)(xr + d);
        float4 wv = *(const float4*)(wr + d);
        s += xv.x * wv.x + xv.y * wv.y + xv.z * wv.z + xv.w * wv.w;
    }
    s += __shfl_xor(s, 16, 32);                      // combine the two half-D partials
    float p = (lane < 16) ? (1.f / (1.f + __expf(-s))) : -3.4e38f;
    #pragma unroll
    for (int it = 0; it < Kk; ++it) {                // iterative argmax over 16 lanes
        float m = p; int mi = lane;
        #pragma unroll
        for (int off = 16; off >= 1; off >>= 1) {
            float om = __shfl_xor(m, off, 32);
            int   oi = __shfl_xor(mi, off, 32);
            if (om > m || (om == m && oi < mi)) { m = om; mi = oi; }
        }
        if (lane == 0) {
            int pos = atomicAdd(&counts[mi], 1);
            lists[mi * Nn + pos] = t;
            glist[mi * Nn + pos] = m;
        }
        if (lane == mi) p = -3.4e38f;
    }
}

// ---------------- fused expert MLP: out += gate * (relu(X W1 + b1) W2 + b2) ----------------
// LDS: sH 32KB (swizzled bf16 H) + sW 2x16KB double-buffered async weight staging = 64KB
__global__ __launch_bounds__(256)
void expert_kernel(const __bf16* __restrict__ xb,      // [N][D] bf16
                   const __bf16* __restrict__ w1t,     // [E][F][D] bf16 (W1 transposed)
                   const __bf16* __restrict__ w2t,     // [E][D][F] bf16 (W2 transposed)
                   const float*  __restrict__ b1,      // [E][F]
                   const float*  __restrict__ b2,      // [E][D]
                   const int*    __restrict__ counts,
                   const int*    __restrict__ lists,
                   const float*  __restrict__ glist,
                   float* __restrict__ out) {          // [N][D]
    __shared__ __bf16 sH[64 * 256];                    // XOR-swizzled H tile
    __shared__ __bf16 sW[2][8192];                     // fragment-ordered weight chunks

    const int e    = blockIdx.y;
    const int cnt  = counts[e];
    const int base = blockIdx.x * 64;
    if (base >= cnt) return;

    const int tid  = threadIdx.x;
    const int w    = tid >> 5, lane = tid & 31;
    const int grp  = lane >> 4, ln = lane & 15;
    const int mt   = w >> 1;          // 0..3 : 16-row slab
    const int nb   = w & 1;           // 0..1 : column half

    const unsigned sWoff0 = (unsigned)(size_t)&sW[0][0];
    const unsigned sWoff1 = (unsigned)(size_t)&sW[1][0];
    const __bf16* w1base = w1t + (size_t)e * Ff * Dd;
    const __bf16* w2base = w2t + (size_t)e * Dd * Ff;

    // ---------- Phase 1: H = relu(X @ W1 + b1) -> sH ----------
    {
        const int ia  = base + mt * 16 + ln;
        const int tokA = (ia < cnt) ? lists[e * Nn + ia] : 0;
        const __bf16* xrow = xb + (size_t)tokA * Dd;

        // stage W1 chunk [256 n][32 k] into sW[buf] in fragment order:
        // chunk c (0..1023, 16B each): t2=c>>6 (n-tile), l=(c>>1)&31 (frag lane), h=c&1
        auto stage1 = [&](int buf, int k0) {
            const unsigned sb = buf ? sWoff1 : sWoff0;
            #pragma unroll
            for (int i = 0; i < 4; ++i) {
                const int c  = i * 256 + tid;
                const int l  = (c >> 1) & 31, h = c & 1, t2 = c >> 6;
                const int n  = t2 * 16 + (l & 15);
                const int kk = k0 + ((l >> 4) << 3) + h * 16;
                async_ld16(sb + c * 16, w1base + (size_t)n * Dd + kk);
            }
        };

        stage1(0, 0);
        v8f acc[8] = {};
        for (int it = 0; it < 32; ++it) {
            if (it + 1 < 32) { stage1((it + 1) & 1, (it + 1) * 32); WAIT_ASYNC(4); }
            else             { WAIT_ASYNC(0); }
            __syncthreads();
            const __bf16* wb = &sW[it & 1][0];
            const int k0 = it * 32;
            // A fragment from global token row
            v8bf alo = *(const v8bf*)(xrow + k0 + grp * 8);
            v8bf ahi = *(const v8bf*)(xrow + k0 + 16 + grp * 8);
            v16bf a;
            #pragma unroll
            for (int i = 0; i < 8; ++i) { a[i] = alo[i]; a[i + 8] = ahi[i]; }
            #pragma unroll
            for (int nt = 0; nt < 8; ++nt) {
                const __bf16* bp = wb + (size_t)(nb * 8 + nt) * 512 + lane * 16;
                v8bf blo = *(const v8bf*)(bp);
                v8bf bhi = *(const v8bf*)(bp + 8);
                v16bf b;
                #pragma unroll
                for (int i = 0; i < 8; ++i) { b[i] = blo[i]; b[i + 8] = bhi[i]; }
                acc[nt] = __builtin_amdgcn_wmma_f32_16x16x32_bf16(
                    false, a, false, b, (short)0, acc[nt], false, false);
            }
            __syncthreads();
        }
        // epilogue: bias + relu, XOR-swizzled store to sH
        #pragma unroll
        for (int nt = 0; nt < 8; ++nt) {
            const int col = nb * 128 + nt * 16 + ln;
            const float bias = b1[(size_t)e * Ff + col];
            #pragma unroll
            for (int v = 0; v < 8; ++v) {
                const int r = mt * 16 + v + 8 * grp;       // C/D layout: M = v + 8*lanegroup
                float h = acc[nt][v] + bias;
                h = h > 0.f ? h : 0.f;
                const int off = r * 256 + ((((col >> 3) ^ (r & 31))) << 3) + (col & 7);
                sH[off] = (__bf16)h;
            }
        }
    }
    __syncthreads();

    // ---------- Phase 2: out += gate * (H @ W2 + b2) ----------
    {
        int tok[8]; float gt[8];
        #pragma unroll
        for (int v = 0; v < 8; ++v) {
            const int idx = base + mt * 16 + v + 8 * grp;
            tok[v] = (idx < cnt) ? lists[e * Nn + idx] : 0;
            gt[v]  = (idx < cnt) ? glist[e * Nn + idx] : 0.f;
        }

        // stage W2 chunk for iteration it: nc=it>>3, k0=(it&7)*32
        auto stage2 = [&](int buf, int it) {
            const unsigned sb = buf ? sWoff1 : sWoff0;
            const int nc = it >> 3, k0 = (it & 7) * 32;
            #pragma unroll
            for (int i = 0; i < 4; ++i) {
                const int c  = i * 256 + tid;
                const int l  = (c >> 1) & 31, h = c & 1, t2 = c >> 6;
                const int n  = (t2 >> 3) * 512 + nc * 128 + (t2 & 7) * 16 + (l & 15);
                const int kk = k0 + ((l >> 4) << 3) + h * 16;
                async_ld16(sb + c * 16, w2base + (size_t)n * Ff + kk);
            }
        };

        stage2(0, 0);
        const int rA = mt * 16 + ln;                      // A row in sH
        for (int nc = 0; nc < 4; ++nc) {
            v8f acc[8] = {};
            for (int ks = 0; ks < 8; ++ks) {
                const int it = nc * 8 + ks;
                if (it + 1 < 32) { stage2((it + 1) & 1, it + 1); WAIT_ASYNC(4); }
                else             { WAIT_ASYNC(0); }
                __syncthreads();
                const __bf16* wb = &sW[it & 1][0];
                // A fragment from swizzled sH (conflict-free ds_load_b128)
                const int galo = (ks * 4 + grp)     ^ (rA & 31);
                const int gahi = (ks * 4 + 2 + grp) ^ (rA & 31);
                v8bf alo = *(const v8bf*)(sH + rA * 256 + galo * 8);
                v8bf ahi = *(const v8bf*)(sH + rA * 256 + gahi * 8);
                v16bf a;
                #pragma unroll
                for (int i = 0; i < 8; ++i) { a[i] = alo[i]; a[i + 8] = ahi[i]; }
                #pragma unroll
                for (int nt = 0; nt < 8; ++nt) {
                    const __bf16* bp = wb + (size_t)(nb * 8 + nt) * 512 + lane * 16;
                    v8bf blo = *(const v8bf*)(bp);
                    v8bf bhi = *(const v8bf*)(bp + 8);
                    v16bf b;
                    #pragma unroll
                    for (int i = 0; i < 8; ++i) { b[i] = blo[i]; b[i + 8] = bhi[i]; }
                    acc[nt] = __builtin_amdgcn_wmma_f32_16x16x32_bf16(
                        false, a, false, b, (short)0, acc[nt], false, false);
                }
                __syncthreads();
            }
            // epilogue: scaled scatter-add (gate==0 rows contribute 0)
            #pragma unroll
            for (int nt = 0; nt < 8; ++nt) {
                const int dcol = nb * 512 + nc * 128 + nt * 16 + ln;
                const float bias = b2[(size_t)e * Dd + dcol];
                #pragma unroll
                for (int v = 0; v < 8; ++v) {
                    const float val = gt[v] * (acc[nt][v] + bias);
                    atomicAdd(out + (size_t)tok[v] * Dd + dcol, val);
                }
            }
        }
    }
}

// ---------------- launch ----------------
extern "C" void kernel_launch(void* const* d_in, const int* in_sizes, int n_in,
                              void* d_out, int out_size, void* d_ws, size_t ws_size,
                              hipStream_t stream) {
    const float* x    = (const float*)d_in[0];
    const float* Wsel = (const float*)d_in[1];
    const float* W1   = (const float*)d_in[2];
    const float* b1   = (const float*)d_in[3];
    const float* W2   = (const float*)d_in[4];
    const float* b2   = (const float*)d_in[5];
    float* out = (float*)d_out;

    char* ws = (char*)d_ws;
    size_t off = 0;
    auto alloc = [&](size_t bytes) -> void* {
        off = (off + 255) & ~(size_t)255;
        void* p = ws + off;
        off += bytes;
        return p;
    };
    int*    counts = (int*)   alloc(Ee * sizeof(int));
    int*    lists  = (int*)   alloc((size_t)Ee * Nn * sizeof(int));
    float*  glist  = (float*) alloc((size_t)Ee * Nn * sizeof(float));
    __bf16* xbf    = (__bf16*)alloc((size_t)Nn * Dd * sizeof(__bf16));
    __bf16* w1t    = (__bf16*)alloc((size_t)Ee * Ff * Dd * sizeof(__bf16));
    __bf16* w2t    = (__bf16*)alloc((size_t)Ee * Dd * Ff * sizeof(__bf16));

    const long n4 = (long)Nn * Dd / 4;                 // 2,097,152

    zero_init<<<(unsigned)((n4 + 255) / 256), 256, 0, stream>>>((float4*)out, counts, n4);
    convert_x<<<(unsigned)((n4 + 255) / 256), 256, 0, stream>>>((const float4*)x, (v4bf*)xbf, n4);

    dim3 tb(32, 8);
    transpose_bf16<<<dim3(Ff / 32, Dd / 32, Ee), tb, 0, stream>>>(W1, w1t, Dd, Ff); // [D,F]->[F,D]
    transpose_bf16<<<dim3(Dd / 32, Ff / 32, Ee), tb, 0, stream>>>(W2, w2t, Ff, Dd); // [F,D]->[D,F]

    route_kernel<<<Nn / 8, 256, 0, stream>>>(x, Wsel, counts, lists, glist);

    expert_kernel<<<dim3(Nn / 64, Ee), 256, 0, stream>>>(xbf, w1t, w2t, b1, b2,
                                                         counts, lists, glist, out);
}
